// Offset_Attention_Position_Embedding_12360915878071
// MI455X (gfx1250) — matrix operand order repaired and checked by reading, not compiled, not run on hardware
//
#include <hip/hip_runtime.h>

// Problem constants (reference: B=8, C=256, N=2048, C4=64, L=4)
#define Bc  8
#define Cc  256
#define Nn  2048
#define C4c 64
#define Lc  4
#define CSCH 8   // n-chunks for attention column-sum reduction

typedef __attribute__((ext_vector_type(16))) __bf16 v16bf;
typedef __attribute__((ext_vector_type(8)))  float  v8f;
typedef __attribute__((ext_vector_type(8)))  __bf16 bf16x8;
typedef __attribute__((ext_vector_type(2)))  __bf16 bf16x2;
typedef __attribute__((ext_vector_type(4)))  unsigned int u32x4;
typedef __attribute__((ext_vector_type(8)))  int i32x8;
typedef __attribute__((ext_vector_type(4)))  int i32x4;

enum { F_B_NK = 1, F_BIAS = 2, F_COLSCALE = 4, F_REVSUB = 8, F_STORE_T = 16 };

// ---------------------------------------------------------------------------
// TDM: DMA a contiguous 16KB block (2048 x 8B) from global memory into LDS.
// D# built per CDNA5 ISA:
//   group0: count=1 | lds_addr | global_addr[56:0] | type=2
//   group1: data_size=3 (8B), tensor_dim0=2048, tensor_dim1=1, tile_dim0=2048,
//           tile_dim1=1, tensor_dim0_stride=2048
// Issued once per wave (EXEC ignored); completion via TENSORcnt.
// ---------------------------------------------------------------------------
__device__ inline void tdm_load_tile_16k(unsigned ldsOff, const void* gptr)
{
    unsigned long long ga = (unsigned long long)(uintptr_t)gptr;
    u32x4 g0;
    g0[0] = 1u;                                                  // count=1, user mode
    g0[1] = ldsOff;                                              // lds_addr (bytes)
    g0[2] = (unsigned)ga;                                        // global_addr[31:0]
    g0[3] = (unsigned)((ga >> 32) & 0x01FFFFFFu) | (2u << 30);   // global_addr[56:32] | type=2
    i32x8 g1;
    g1[0] = 0x00030000;   // workgroup_mask=0, data_size=3 (8B)
    g1[1] = 0x08000000;   // tensor_dim0 = 2048  (bits[79:48], low 16 here)
    g1[2] = 0x00010000;   // tensor_dim1 = 1     (bits[111:80])
    g1[3] = 0x08000000;   // tile_dim0 = 2048    (bits[127:112])
    g1[4] = 1;            // tile_dim1 = 1, tile_dim2 = 0
    g1[5] = 2048;         // tensor_dim0_stride = 2048
    g1[6] = 0;
    g1[7] = 0;
    i32x4 z4 = {0, 0, 0, 0};
#if __has_include(<hip/amd_detail/amd_gfx1250_TDM.h>)
    i32x8 z8 = {0, 0, 0, 0, 0, 0, 0, 0};
    __builtin_amdgcn_tensor_load_to_lds(g0, g1, z4, z4, z8, 0);
#else
    __builtin_amdgcn_tensor_load_to_lds(g0, g1, z4, z4, 0);
#endif
}

// ---------------------------------------------------------------------------
// Energy GEMM: att[b][n][m] = sum_d q[b][n][d]*q[b][m][d], q is bf16 [N][64].
// Block tile 128x128, K=64 staged entirely by two TDM loads (16KB each).
// 8 waves (4x2), wave tile 32x64 -> 16 WMMAs per wave.
// ---------------------------------------------------------------------------
__global__ __launch_bounds__(256)
void energy_wmma_tdm(const __bf16* __restrict__ q, float* __restrict__ att)
{
    constexpr int NN = 2048;
    __shared__ __align__(16) __bf16 As[128 * 64];
    __shared__ __align__(16) __bf16 Bs[128 * 64];

    const int tid = threadIdx.x, lane = tid & 31, wave = tid >> 5;
    const int wm = (wave % 4) * 32, wn = (wave / 4) * 64;
    const int m0 = blockIdx.y * 128, n0 = blockIdx.x * 128, bz = blockIdx.z;
    const __bf16* qb = q + (long)bz * NN * 64;

    if (wave == 0) {
        tdm_load_tile_16k((unsigned)(uintptr_t)(void*)As, qb + (long)m0 * 64);
        tdm_load_tile_16k((unsigned)(uintptr_t)(void*)Bs, qb + (long)n0 * 64);
        __builtin_amdgcn_s_wait_tensorcnt(0);
    }
    __syncthreads();

    const int half = lane >> 4, ml = lane & 15;
    v8f acc[2][4] = {};
    #pragma unroll
    for (int ks = 0; ks < 2; ks++) {
        v16bf af[2], bfv[4];
        #pragma unroll
        for (int i = 0; i < 2; i++) {
            const __bf16* row = As + (wm + i * 16 + ml) * 64 + ks * 32;
            bf16x8 lo = *(const bf16x8*)(row + 8 * half);
            bf16x8 hi = *(const bf16x8*)(row + 16 + 8 * half);
            #pragma unroll
            for (int e = 0; e < 8; e++) { af[i][e] = lo[e]; af[i][e + 8] = hi[e]; }
        }
        #pragma unroll
        for (int j = 0; j < 4; j++) {
            const __bf16* col = Bs + (wn + j * 16 + ml) * 64 + ks * 32;
            bf16x8 lo = *(const bf16x8*)(col + 8 * half);
            bf16x8 hi = *(const bf16x8*)(col + 16 + 8 * half);
            #pragma unroll
            for (int e = 0; e < 8; e++) { bfv[j][e] = lo[e]; bfv[j][e + 8] = hi[e]; }
        }
        #pragma unroll
        for (int i = 0; i < 2; i++)
            #pragma unroll
            for (int j = 0; j < 4; j++)
                acc[i][j] = __builtin_amdgcn_wmma_f32_16x16x32_bf16(
                    false, af[i], false, bfv[j], (short)0, acc[i][j], false, false);
    }

    float* Ob = att + (long)bz * NN * NN;
    #pragma unroll
    for (int i = 0; i < 2; i++) {
        #pragma unroll
        for (int j = 0; j < 4; j++) {
            const int ng  = n0 + wn + j * 16 + ml;
            const int mg0 = m0 + wm + i * 16 + 8 * half;
            float* obase = Ob + (long)mg0 * NN + ng;
            #pragma unroll
            for (int r = 0; r < 8; r++) obase[r * NN] = acc[i][j][r];
        }
    }
}

// ---------------------------------------------------------------------------
// Generic tiled GEMM: Out[b] = A[b](MxK, row-major, f32) * B[b](K x 2048, f32)
//   bf16 operands double-buffered through LDS, f32 accum, v_wmma_f32_16x16x32_bf16.
//   Block tile BM x 128, BK=32, 256 threads = 8 waves.
//     BM=128: waves 4x2, wave tile 32x64 -> 8 wmma / k-step / wave
//     BM= 64: waves 2x4, wave tile 32x32 -> 4 wmma / k-step / wave (qk GEMM, M=64)
//   Software pipeline: ds_load frags -> global prefetch next -> wmma -> cvt+ds_store.
//   FLAGS: F_BIAS add bias[m]; F_COLSCALE scale by colscale[b][n];
//          F_REVSUB Out = Res - result; F_STORE_T store bf16 transposed [n][m].
// ---------------------------------------------------------------------------
template<int FLAGS, int BM>
__global__ __launch_bounds__(256)
void gemm_bf16_wmma(const float* __restrict__ A, long aBatch,
                    const float* __restrict__ Bsrc, long bBatch,
                    const float* __restrict__ bias,
                    const float* __restrict__ colscale,
                    const float* __restrict__ Res, long resBatch,
                    float* __restrict__ Out, long oBatch,
                    int M, int K)
{
    constexpr int NN    = 2048;
    constexpr int WROWS = (BM == 128) ? 4 : 2;
    constexpr int WCOLS = 8 / WROWS;
    constexpr int WTN   = 128 / WCOLS;
    constexpr int FN    = WTN / 16;
    constexpr int EA    = BM / 8;

    __shared__ __align__(16) __bf16 As[2][BM * 32];
    __shared__ __align__(16) __bf16 Bs[2][128 * 32];

    const int tid  = threadIdx.x;
    const int lane = tid & 31;
    const int wave = tid >> 5;
    const int wm   = (wave % WROWS) * 32;
    const int wn   = (wave / WROWS) * WTN;
    const int m0   = blockIdx.y * BM;
    const int n0   = blockIdx.x * 128;
    const int bz   = blockIdx.z;

    const float* Ab = A    + (long)bz * aBatch;
    const float* Bb = Bsrc + (long)bz * bBatch;

    float ra[EA];
    float rb[16];

    auto gload = [&](int k0) {
        {
            const int t = tid * EA, r = t >> 5, kc = t & 31;
            const float* src = Ab + (long)(m0 + r) * K + k0 + kc;
            #pragma unroll
            for (int q = 0; q < EA; q += 4) {
                float4 f = *(const float4*)(src + q);
                ra[q] = f.x; ra[q + 1] = f.y; ra[q + 2] = f.z; ra[q + 3] = f.w;
            }
        }
        {
            const int n = tid & 127, kh = (tid >> 7) * 16;
            const float* src = Bb + (long)(k0 + kh) * NN + n0 + n;
            #pragma unroll
            for (int kk = 0; kk < 16; kk++) rb[kk] = src[kk * NN];
        }
    };
    auto lstore = [&](int buf) {
        {
            const int t = tid * EA, r = t >> 5, kc = t & 31;
            __bf16* d = &As[buf][r * 32 + kc];
            #pragma unroll
            for (int q = 0; q < EA; q += 8) {
                bf16x8 v;
                #pragma unroll
                for (int e = 0; e < 8; e++) v[e] = (__bf16)ra[q + e];
                *(bf16x8*)(d + q) = v;
            }
        }
        {
            const int n = tid & 127, kh = (tid >> 7) * 16;
            __bf16* d = &Bs[buf][n * 32 + kh];
            #pragma unroll
            for (int kk = 0; kk < 16; kk += 2) {
                bf16x2 p; p[0] = (__bf16)rb[kk]; p[1] = (__bf16)rb[kk + 1];
                *(bf16x2*)(d + kk) = p;
            }
        }
    };

    v8f acc[2][FN] = {};
    const int half = lane >> 4;
    const int ml   = lane & 15;

    gload(0);
    lstore(0);
    const int nk = K >> 5;
    for (int ks = 0; ks < nk; ks++) {
        __syncthreads();
        const __bf16* Acur = As[ks & 1];
        const __bf16* Bcur = Bs[ks & 1];

        v16bf af[2], bfv[FN];
        #pragma unroll
        for (int i = 0; i < 2; i++) {
            const __bf16* row = Acur + (wm + i * 16 + ml) * 32;
            bf16x8 lo = *(const bf16x8*)(row + 8 * half);
            bf16x8 hi = *(const bf16x8*)(row + 16 + 8 * half);
            #pragma unroll
            for (int e = 0; e < 8; e++) { af[i][e] = lo[e]; af[i][e + 8] = hi[e]; }
        }
        #pragma unroll
        for (int j = 0; j < FN; j++) {
            const __bf16* col = Bcur + (wn + j * 16 + ml) * 32;
            bf16x8 lo = *(const bf16x8*)(col + 8 * half);
            bf16x8 hi = *(const bf16x8*)(col + 16 + 8 * half);
            #pragma unroll
            for (int e = 0; e < 8; e++) { bfv[j][e] = lo[e]; bfv[j][e + 8] = hi[e]; }
        }

        if (ks + 1 < nk) gload((ks + 1) << 5);   // prefetch next tile under the WMMAs

        #pragma unroll
        for (int i = 0; i < 2; i++)
            #pragma unroll
            for (int j = 0; j < FN; j++)
                acc[i][j] = __builtin_amdgcn_wmma_f32_16x16x32_bf16(
                    false, af[i], false, bfv[j], (short)0, acc[i][j], false, false);

        if (ks + 1 < nk) lstore((ks + 1) & 1);   // fill the other LDS buffer
    }

    // ---- epilogue ----
    const float* Rb  = (FLAGS & F_REVSUB)   ? (Res + (long)bz * resBatch) : nullptr;
    const float* CSb = (FLAGS & F_COLSCALE) ? (colscale + (long)bz * NN)  : nullptr;

    #pragma unroll
    for (int i = 0; i < 2; i++) {
        #pragma unroll
        for (int j = 0; j < FN; j++) {
            const int ng  = n0 + wn + j * 16 + ml;
            const int mg0 = m0 + wm + i * 16 + 8 * half;
            float cscale = 1.0f;
            if (FLAGS & F_COLSCALE) cscale = CSb[ng];
            if (FLAGS & F_STORE_T) {
                // bf16 transposed store: rows r -> consecutive elements, one b128
                __bf16* ObT = (__bf16*)Out + (long)bz * oBatch + (long)ng * M + mg0;
                bf16x8 v;
                #pragma unroll
                for (int e = 0; e < 8; e++) v[e] = (__bf16)acc[i][j][e];
                *(bf16x8*)ObT = v;
            } else {
                float* obase = Out + (long)bz * oBatch + (long)mg0 * NN + ng;
                const float* rbase = (FLAGS & F_REVSUB) ? (Rb + (long)mg0 * NN + ng) : nullptr;
                #pragma unroll
                for (int r = 0; r < 8; r++) {       // r*NN folds into immediate offsets
                    float v = acc[i][j][r];
                    if (FLAGS & F_COLSCALE) v *= cscale;
                    if (FLAGS & F_BIAS)     v += bias[mg0 + r];
                    if (FLAGS & F_REVSUB)   v = rbase[r * NN] - v;
                    obase[r * NN] = v;
                }
            }
        }
    }
}

// ---------------------------------------------------------------------------
__global__ __launch_bounds__(256)
void pos_embed_kernel(const float* __restrict__ xyz, const float* __restrict__ pw,
                      const float* __restrict__ pb, float* __restrict__ xe)
{
    const int n = blockIdx.x * 256 + threadIdx.x;
    const int c = blockIdx.y, b = blockIdx.z;
    const float* p = xyz + ((long)b * Nn + n) * 3;
    const float v = pw[c * 3 + 0] * p[0] + pw[c * 3 + 1] * p[1] + pw[c * 3 + 2] * p[2] + pb[c];
    xe[((long)b * Cc + c) * Nn + n] = v;
}

__global__ __launch_bounds__(256)
void add2_kernel(const float* __restrict__ a, const float* __restrict__ b,
                 float* __restrict__ o)
{
    const long i = (long)blockIdx.x * 256 + threadIdx.x;
    o[i] = a[i] + b[i];
}

// row softmax in place; row length 2048, 8 elems/thread
__global__ __launch_bounds__(256)
void softmax_rows_kernel(float* __restrict__ att)
{
    float* row = att + ((long)blockIdx.y * Nn + blockIdx.x) * Nn;
    const int tid = threadIdx.x;
    float vals[8];
    float mx = -3.402823e38f;
    #pragma unroll
    for (int j = 0; j < 8; j++) { vals[j] = row[tid + j * 256]; mx = fmaxf(mx, vals[j]); }
    __shared__ float red[256];
    red[tid] = mx; __syncthreads();
    for (int s = 128; s > 0; s >>= 1) { if (tid < s) red[tid] = fmaxf(red[tid], red[tid + s]); __syncthreads(); }
    mx = red[0]; __syncthreads();
    float sum = 0.f;
    #pragma unroll
    for (int j = 0; j < 8; j++) { vals[j] = __expf(vals[j] - mx); sum += vals[j]; }
    red[tid] = sum; __syncthreads();
    for (int s = 128; s > 0; s >>= 1) { if (tid < s) red[tid] += red[tid + s]; __syncthreads(); }
    const float inv = 1.0f / red[0];
    #pragma unroll
    for (int j = 0; j < 8; j++) row[tid + j * 256] = vals[j] * inv;
}

__global__ __launch_bounds__(256)
void colsum_part_kernel(const float* __restrict__ att, float* __restrict__ part)
{
    const int m = blockIdx.x * 256 + threadIdx.x;
    const int b = blockIdx.y, ch = blockIdx.z;
    const float* base = att + (long)b * Nn * Nn;
    const int n0 = ch * (Nn / CSCH);
    float s = 0.f;
    for (int n = n0; n < n0 + Nn / CSCH; n++) s += base[(long)n * Nn + m];
    part[((long)ch * Bc + b) * Nn + m] = s;
}

__global__ __launch_bounds__(256)
void colsum_finish_kernel(const float* __restrict__ part, float* __restrict__ cs)
{
    const int m = blockIdx.x * 256 + threadIdx.x;
    const int b = blockIdx.y;
    float s = 0.f;
    #pragma unroll
    for (int ch = 0; ch < CSCH; ch++) s += part[((long)ch * Bc + b) * Nn + m];
    cs[(long)b * Nn + m] = 1.0f / (1e-9f + s);
}

// per-channel training-mode BN stats over (B, N): a = gamma*rsqrt(var+eps), b = beta - mean*a
__global__ __launch_bounds__(256)
void bn_stats_kernel(const float* __restrict__ y, const float* __restrict__ gamma,
                     const float* __restrict__ beta, float* __restrict__ bnA,
                     float* __restrict__ bnB)
{
    const int c = blockIdx.x, tid = threadIdx.x;
    float s = 0.f, sq = 0.f;
    for (int b = 0; b < Bc; b++) {
        const float* p = y + ((long)b * Cc + c) * Nn;
        for (int n = tid; n < Nn; n += 256) { const float v = p[n]; s += v; sq += v * v; }
    }
    __shared__ float rs[256], rq[256];
    rs[tid] = s; rq[tid] = sq; __syncthreads();
    for (int st = 128; st > 0; st >>= 1) {
        if (tid < st) { rs[tid] += rs[tid + st]; rq[tid] += rq[tid + st]; }
        __syncthreads();
    }
    if (tid == 0) {
        const float cnt  = (float)(Bc * Nn);
        const float mean = rs[0] / cnt;
        const float var  = rq[0] / cnt - mean * mean;
        const float a    = gamma[c] * rsqrtf(var + 1e-5f);
        bnA[c] = a;
        bnB[c] = beta[c] - mean * a;
    }
}

// dst1 = (res? res:0) + relu(a[c]*y + b[c]); optionally mirror to out concat slice
__global__ __launch_bounds__(256)
void bn_apply_kernel(const float* __restrict__ y, const float* __restrict__ bnA,
                     const float* __restrict__ bnB, const float* __restrict__ res,
                     float* __restrict__ dst1, float* __restrict__ dst2, int chanOff)
{
    const int n = blockIdx.x * 256 + threadIdx.x;
    const int c = blockIdx.y, b = blockIdx.z;
    const long idx = ((long)b * Cc + c) * Nn + n;
    float v = fmaxf(bnA[c] * y[idx] + bnB[c], 0.0f);
    if (res) v += res[idx];
    dst1[idx] = v;
    if (dst2) dst2[((long)b * (4 * Cc) + chanOff + c) * Nn + n] = v;
}

// ---------------------------------------------------------------------------
extern "C" void kernel_launch(void* const* d_in, const int* in_sizes, int n_in,
                              void* d_out, int out_size, void* d_ws, size_t ws_size,
                              hipStream_t stream)
{
    (void)in_sizes; (void)n_in; (void)out_size; (void)ws_size;
    const float* x       = (const float*)d_in[0];
    const float* xyz     = (const float*)d_in[1];
    const float* conv1_w = (const float*)d_in[2];
    const float* pos_w   = (const float*)d_in[3];
    const float* pos_b   = (const float*)d_in[4];
    const float* bn1_g   = (const float*)d_in[5];
    const float* bn1_b   = (const float*)d_in[6];
    const float* qk_w    = (const float*)d_in[7];   // [L][64][256]
    const float* v_w     = (const float*)d_in[8];   // [L][256][256]
    const float* v_b     = (const float*)d_in[9];
    const float* trans_w = (const float*)d_in[10];
    const float* trans_b = (const float*)d_in[11];
    const float* oa_g    = (const float*)d_in[12];
    const float* oa_b    = (const float*)d_in[13];
    float* out = (float*)d_out;

    const long BCN = (long)Bc * Cc * Nn;
    const long CN  = (long)Cc * Nn;
    float* ws = (float*)d_ws;
    long off = 0;
    float* xyz_e = ws + off; off += BCN;
    float* hprev = ws + off; off += BCN;
    float* hbuf  = ws + off; off += BCN;
    float* ybuf  = ws + off; off += BCN;
    float* xv    = ws + off; off += BCN;
    float* ubuf  = ws + off; off += BCN;
    float* qbuf  = ws + off; off += (long)Bc * Nn * C4c / 2;  // bf16 [B][N][64]
    float* cs    = ws + off; off += (long)Bc * Nn;
    float* cspart= ws + off; off += (long)CSCH * Bc * Nn;
    float* bnA   = ws + off; off += Cc;
    float* bnB   = ws + off; off += Cc;
    float* att   = ws + off; off += (long)Bc * Nn * Nn;       // energy / att in place

    const dim3 blk(256);

    // positional embedding
    pos_embed_kernel<<<dim3(Nn / 256, Cc, Bc), blk, 0, stream>>>(xyz, pos_w, pos_b, xyz_e);

    // h0 = relu(bn1(conv1_w @ x))
    gemm_bf16_wmma<0, 128><<<dim3(Nn / 128, Cc / 128, Bc), blk, 0, stream>>>(
        conv1_w, 0, x, CN, nullptr, nullptr, nullptr, 0, ybuf, CN, Cc, Cc);
    bn_stats_kernel<<<dim3(Cc), blk, 0, stream>>>(ybuf, bn1_g, bn1_b, bnA, bnB);
    bn_apply_kernel<<<dim3(Nn / 256, Cc, Bc), blk, 0, stream>>>(
        ybuf, bnA, bnB, nullptr, hprev, nullptr, 0);

    for (int i = 0; i < Lc; i++) {
        // h = h_prev + xyz_e
        add2_kernel<<<dim3((unsigned)(BCN / 256)), blk, 0, stream>>>(hprev, xyz_e, hbuf);

        // q[b][n][d] = sum_c qk_w[i][d][c] h[b][c][n]   (bf16, stored transposed)
        gemm_bf16_wmma<F_STORE_T, 64><<<dim3(Nn / 128, 1, Bc), blk, 0, stream>>>(
            qk_w + (long)i * C4c * Cc, 0, hbuf, CN, nullptr, nullptr, nullptr, 0,
            qbuf, (long)Nn * C4c /*bf16 elems*/, C4c, Cc);

        // xv = v_w @ h + v_b
        gemm_bf16_wmma<F_BIAS, 128><<<dim3(Nn / 128, Cc / 128, Bc), blk, 0, stream>>>(
            v_w + (long)i * Cc * Cc, 0, hbuf, CN, v_b + (long)i * Cc, nullptr, nullptr, 0,
            xv, CN, Cc, Cc);

        // energy[n][m] = q[n][:] . q[m][:]   (TDM-staged, bf16 operands)
        energy_wmma_tdm<<<dim3(Nn / 128, Nn / 128, Bc), blk, 0, stream>>>(
            (const __bf16*)qbuf, att);

        // att = softmax rows; cs = 1/(1e-9 + colsum)
        softmax_rows_kernel<<<dim3(Nn, Bc), blk, 0, stream>>>(att);
        colsum_part_kernel<<<dim3(Nn / 256, Bc, CSCH), blk, 0, stream>>>(att, cspart);
        colsum_finish_kernel<<<dim3(Nn / 256, Bc), blk, 0, stream>>>(cspart, cs);

        // u = h - (xv @ att) * cs   (column renorm + offset folded into epilogue)
        gemm_bf16_wmma<F_COLSCALE | F_REVSUB, 128><<<dim3(Nn / 128, Cc / 128, Bc), blk, 0, stream>>>(
            xv, CN, att, (long)Nn * Nn, nullptr, cs, hbuf, CN, ubuf, CN, Cc, Nn);

        // y = trans_w @ u + trans_b
        gemm_bf16_wmma<F_BIAS, 128><<<dim3(Nn / 128, Cc / 128, Bc), blk, 0, stream>>>(
            trans_w + (long)i * Cc * Cc, 0, ubuf, CN, trans_b + (long)i * Cc, nullptr, nullptr, 0,
            ybuf, CN, Cc, Cc);

        // h_out = h + relu(bn(y)); mirror to out[:, i*C:(i+1)*C, :]
        bn_stats_kernel<<<dim3(Cc), blk, 0, stream>>>(
            ybuf, oa_g + (long)i * Cc, oa_b + (long)i * Cc, bnA, bnB);
        bn_apply_kernel<<<dim3(Nn / 256, Cc, Bc), blk, 0, stream>>>(
            ybuf, bnA, bnB, hbuf, hprev, out, i * Cc);
    }
}
